// gs_sum_concatenation_gs_26328149524903
// MI455X (gfx1250) — compile-verified, matrix operand order checked by reading, not run on hardware
//
#include <hip/hip_runtime.h>

// ---------------------------------------------------------------------------
// CDNA5 / gfx1250 implementation. wave32 everywhere. Dense layers use
// v_wmma_f32_16x16x32_bf16 (f32 accumulate) with double-buffered LDS tiles and
// async global->LDS staging (ASYNCcnt); edge ops are wave-per-edge with
// L2-resident float atomics.
// ---------------------------------------------------------------------------

typedef __attribute__((ext_vector_type(16))) __bf16 v16bf;
typedef __attribute__((ext_vector_type(8)))  float  v8f;
typedef __attribute__((ext_vector_type(8)))  unsigned short us8;

union FragU { us8 u[2]; v16bf v; };

#define NA_  50000
#define NP_  100000
#define FA_  19
#define LAT_ 128
#define EC_  800000
#define EA_  400000
#define EAP_ 250000
#define ELP_ 200000

__device__ __forceinline__ unsigned short f2bf(float f) {
  unsigned int u = __float_as_uint(f);
  u += 0x7FFFu + ((u >> 16) & 1u);           // round-to-nearest-even
  return (unsigned short)(u >> 16);
}
__device__ __forceinline__ float sigf(float x) { return 1.f / (1.f + __expf(-x)); }

__device__ __forceinline__ void atomicMaxF(float* addr, float val) {
  unsigned int* a = (unsigned int*)addr;
  unsigned int old = *a;
  for (;;) {
    if (__uint_as_float(old) >= val) return;
    unsigned int prev = atomicCAS(a, old, __float_as_uint(val));
    if (prev == old) return;
    old = prev;
  }
}

// gfx1250 async global->LDS copy, 16B per lane (ASYNCcnt-tracked, no VGPR
// round trip). lds_off = byte offset of destination in LDS.
__device__ __forceinline__ void async_ld_b128(unsigned lds_off, const void* gaddr) {
  asm volatile("global_load_async_to_lds_b128 %0, %1, off"
               :: "v"(lds_off), "v"(gaddr) : "memory");
}
__device__ __forceinline__ void wait_async0() {
  asm volatile("s_wait_asynccnt 0" ::: "memory");
}

// ---------------------------------------------------------------------------
// Generic bf16 WMMA GEMM: C[M x N] = act(A[M x Kp](bf16) @ B[K x N](bf16) + bias)
// Block tile 128x128, 8 waves, each wave a 64x32 sub-tile = 4x2 WMMA tiles.
// Double-buffered LDS: A tile row-major (async-staged for interior M blocks);
// B tile staged transposed ([col][k]) so each lane's fragment is two
// contiguous 16B LDS loads, matching the ISA 16-bit A/B VGPR layouts
// (lanes<16 hold K {kg..kg+7, kg+16..kg+23} with kg=0; lanes>=16 kg=8).
// ---------------------------------------------------------------------------
__global__ __launch_bounds__(256)
void wmma_gemm_kernel(const unsigned short* __restrict__ A, int Kp,
                      const unsigned short* __restrict__ B, int ldb, int K,
                      const float* __restrict__ bias,
                      float* __restrict__ C, int ldc,
                      int Mrows, int Ncols, int relu) {
  __shared__ unsigned short sA[2][128 * 32];   // [buf][row][k]
  __shared__ unsigned short sB[2][128 * 32];   // [buf][col][k] (transposed)
  const int tid  = threadIdx.x;
  const int lane = tid & 31;
  const int wave = tid >> 5;
  const int wm = (wave & 1) * 64;           // wave row offset inside tile
  const int wn = (wave >> 1) * 32;          // wave col offset inside tile
  const int bm = blockIdx.y * 128;
  const int bn = blockIdx.x * 128;
  const int l15 = lane & 15;
  const int kg  = (lane >> 4) * 8;          // 0 or 8
  const bool interiorM = (bm + 128 <= Mrows);

  auto stageA = [&](int kt, int b) {
#pragma unroll
    for (int i = 0; i < 2; ++i) {
      int v  = tid + i * 256;               // 0..511
      int r  = v >> 2;                      // 0..127
      int cv = (v & 3) * 8;                 // 0,8,16,24
      if (interiorM) {
        unsigned lds = (unsigned)(size_t)(const void*)&sA[b][r * 32 + cv];
        async_ld_b128(lds, (const void*)(A + (size_t)(bm + r) * Kp + kt + cv));
      } else {
        us8 val;
        if (bm + r < Mrows) {
          val = *(const us8*)(A + (size_t)(bm + r) * Kp + kt + cv);
        } else {
#pragma unroll
          for (int q = 0; q < 8; ++q) val[q] = 0;
        }
        *(us8*)(&sA[b][r * 32 + cv]) = val;
      }
    }
  };
  auto stageB = [&](int kt, int b) {
#pragma unroll
    for (int i = 0; i < 16; ++i) {
      int idx = tid + i * 256;              // 0..4095
      int k   = idx >> 7;                   // 0..31
      int col = idx & 127;
      unsigned short val = 0;
      if ((kt + k) < K && (bn + col) < Ncols)
        val = B[(size_t)(kt + k) * ldb + bn + col];
      sB[b][col * 32 + k] = val;
    }
  };

  v8f acc[4][2];
#pragma unroll
  for (int mi = 0; mi < 4; ++mi)
#pragma unroll
    for (int ni = 0; ni < 2; ++ni)
#pragma unroll
      for (int q = 0; q < 8; ++q) acc[mi][ni][q] = 0.f;

  // prologue: stage first K-tile
  stageA(0, 0);
  stageB(0, 0);
  wait_async0();
  __syncthreads();

  const int nk = Kp >> 5;
  for (int it = 0; it < nk; ++it) {
    const int cur = it & 1;
    const int nxt = cur ^ 1;

    // fragment loads from current buffer (issued before next-tile staging so
    // the dscnt wait for them does not cover the staging stores)
    FragU af[4], bf[2];
#pragma unroll
    for (int mi = 0; mi < 4; ++mi) {
      const unsigned short* base = &sA[cur][(wm + mi * 16 + l15) * 32 + kg];
      af[mi].u[0] = *(const us8*)base;
      af[mi].u[1] = *(const us8*)(base + 16);
    }
#pragma unroll
    for (int ni = 0; ni < 2; ++ni) {
      const unsigned short* base = &sB[cur][(wn + ni * 16 + l15) * 32 + kg];
      bf[ni].u[0] = *(const us8*)base;
      bf[ni].u[1] = *(const us8*)(base + 16);
    }

    // stage next K-tile into the other buffer (overlaps the WMMAs below)
    if (it + 1 < nk) {
      int ktn = (it + 1) * 32;
      stageA(ktn, nxt);
      stageB(ktn, nxt);
      if (it + 2 < nk) {   // warm L2 for the tile after next
        int pr = bm + (tid >> 1);
        if (pr < Mrows)
          __builtin_prefetch(A + (size_t)pr * Kp + (it + 2) * 32 + (tid & 1) * 16, 0, 1);
      }
    }

#pragma unroll
    for (int mi = 0; mi < 4; ++mi)
#pragma unroll
      for (int ni = 0; ni < 2; ++ni)
        acc[mi][ni] = __builtin_amdgcn_wmma_f32_16x16x32_bf16(
            false, af[mi].v, false, bf[ni].v, (short)0, acc[mi][ni], false, false);

    wait_async0();
    __syncthreads();
  }

  // ---- epilogue: C 16x16 f32 layout: VGPR i, lane L -> row i + (L>=16?8:0),
  //      col L&15 ----
  const int rhi = (lane >> 4) * 8;
#pragma unroll
  for (int mi = 0; mi < 4; ++mi) {
#pragma unroll
    for (int ni = 0; ni < 2; ++ni) {
      int colg = bn + wn + ni * 16 + l15;
#pragma unroll
      for (int i = 0; i < 8; ++i) {
        int rowg = bm + wm + mi * 16 + i + rhi;
        if (rowg < Mrows && colg < Ncols) {
          float v = acc[mi][ni][i] + (bias ? bias[colg] : 0.f);
          if (relu) v = fmaxf(v, 0.f);
          C[(size_t)rowg * ldc + colg] = v;
        }
      }
    }
  }
}

// f32 -> bf16 with optional K padding (Kp >= K, pad = 0)
__global__ void cvt_bf16_kernel(const float* __restrict__ src,
                                unsigned short* __restrict__ dst,
                                long rows, int K, int Kp) {
  long i = blockIdx.x * (long)blockDim.x + threadIdx.x;
  long total = rows * (long)Kp;
  if (i >= total) return;
  long r = i / Kp;
  int  c = (int)(i % Kp);
  dst[i] = (c < K) ? f2bf(src[r * (long)K + c]) : (unsigned short)0;
}

__global__ void fill_f32_kernel(float* p, float v, long n) {
  long i = blockIdx.x * (long)blockDim.x + threadIdx.x;
  if (i < n) p[i] = v;
}

// ---------------------------------------------------------------------------
// GATv2 edge kernels (edges + N appended self-loops).
// ---------------------------------------------------------------------------
__global__ __launch_bounds__(256)
void gat_edge_score_kernel(const float* __restrict__ xl, const float* __restrict__ xr,
                           const float* __restrict__ att,
                           const int* __restrict__ src, const int* __restrict__ dst,
                           int E, int N, float* __restrict__ esc,
                           float* __restrict__ emax) {
  int e = blockIdx.x * 8 + (threadIdx.x >> 5);
  int lane = threadIdx.x & 31;
  if (e >= E + N) return;
  int s = (e < E) ? src[e] : (e - E);
  int d = (e < E) ? dst[e] : (e - E);
  float sum = 0.f;
#pragma unroll
  for (int j = 0; j < 4; ++j) {
    int c = lane + 32 * j;
    float v = xl[(size_t)s * 128 + c] + xr[(size_t)d * 128 + c];
    v = (v > 0.f) ? v : 0.2f * v;           // LeakyReLU(0.2)
    sum += v * att[c];
  }
  for (int off = 16; off; off >>= 1) sum += __shfl_xor(sum, off, 32);
  if (lane == 0) { esc[e] = sum; atomicMaxF(&emax[d], sum); }
}

__global__ void gat_edge_exp_kernel(const float* __restrict__ esc,
                                    const float* __restrict__ emax,
                                    const int* __restrict__ dst, int E, int N,
                                    float* __restrict__ eex, float* __restrict__ denom) {
  int e = blockIdx.x * blockDim.x + threadIdx.x;
  if (e >= E + N) return;
  int d = (e < E) ? dst[e] : (e - E);
  float x = __expf(esc[e] - emax[d]);
  eex[e] = x;
  atomicAdd(&denom[d], x);
}

__global__ __launch_bounds__(256)
void gat_scatter_kernel(const float* __restrict__ xl, const float* __restrict__ eex,
                        const float* __restrict__ denom,
                        const int* __restrict__ src, const int* __restrict__ dst,
                        int E, int N, float* __restrict__ out) {
  int e = blockIdx.x * 8 + (threadIdx.x >> 5);
  int lane = threadIdx.x & 31;
  if (e >= E + N) return;
  int s = (e < E) ? src[e] : (e - E);
  int d = (e < E) ? dst[e] : (e - E);
  float alpha = eex[e] / denom[d];
#pragma unroll
  for (int j = 0; j < 4; ++j) {
    int c = lane + 32 * j;
    atomicAdd(&out[(size_t)d * 128 + c], alpha * xl[(size_t)s * 128 + c]);
  }
}

__global__ void bias_act_kernel(float* __restrict__ h, const float* __restrict__ bias,
                                long rows, int cols, int relu) {
  long i = blockIdx.x * (long)blockDim.x + threadIdx.x;
  if (i >= rows * cols) return;
  int c = (int)(i % cols);
  float v = h[i] + bias[c];
  h[i] = relu ? fmaxf(v, 0.f) : v;
}

// collab[author] += h_paper[paper], 128-wide, wave per edge
__global__ __launch_bounds__(256)
void segsum_kernel(const float* __restrict__ hp, const int* __restrict__ ap_paper,
                   const int* __restrict__ ap_author, int E, float* __restrict__ out) {
  int e = blockIdx.x * 8 + (threadIdx.x >> 5);
  int lane = threadIdx.x & 31;
  if (e >= E) return;
  int p = ap_paper[e], a = ap_author[e];
#pragma unroll
  for (int j = 0; j < 4; ++j) {
    int c = lane + 32 * j;
    atomicAdd(&out[(size_t)a * 128 + c], hp[(size_t)p * 128 + c]);
  }
}

__global__ void concat_kernel(const float* __restrict__ xa, const float* __restrict__ collab,
                              float* __restrict__ xcat, int Na) {
  long i = blockIdx.x * (long)blockDim.x + threadIdx.x;
  long total = (long)Na * 147;
  if (i >= total) return;
  long r = i / 147;
  int  c = (int)(i % 147);
  xcat[i] = (c < 19) ? xa[r * 19 + c] : collab[r * 128 + (c - 19)];
}

// gates = gx + gh per gate block (i,f,g,o); update c,h
__global__ void lstm_cell_kernel(const float* __restrict__ gx, const float* __restrict__ gh,
                                 float* __restrict__ c, float* __restrict__ h, int N) {
  long i = blockIdx.x * (long)blockDim.x + threadIdx.x;
  if (i >= (long)N * 128) return;
  long n = i >> 7;
  int  k = (int)(i & 127);
  const float* gxr = gx + n * 512;
  const float* ghr = gh + n * 512;
  float ig = sigf(gxr[k] + ghr[k]);
  float fg = sigf(gxr[128 + k] + ghr[128 + k]);
  float gg = tanhf(gxr[256 + k] + ghr[256 + k]);
  float og = sigf(gxr[384 + k] + ghr[384 + k]);
  float cn = fg * c[i] + ig * gg;
  c[i] = cn;
  h[i] = og * tanhf(cn);
}

// out[e] = dot(z[src]*z[dst], W) + b   (wave per edge)
__global__ __launch_bounds__(256)
void link_head_kernel(const float* __restrict__ z, const int* __restrict__ src,
                      const int* __restrict__ dst, const float* __restrict__ W,
                      const float* __restrict__ b, float* __restrict__ out, int E) {
  int e = blockIdx.x * 8 + (threadIdx.x >> 5);
  int lane = threadIdx.x & 31;
  if (e >= E) return;
  size_t s = (size_t)src[e], d = (size_t)dst[e];
  float sum = 0.f;
#pragma unroll
  for (int j = 0; j < 4; ++j) {
    int c = lane + 32 * j;
    sum += z[s * 128 + c] * z[d * 128 + c] * W[c];
  }
  for (int off = 16; off; off >>= 1) sum += __shfl_xor(sum, off, 32);
  if (lane == 0) out[e] = sum + b[0];
}

// ---------------------------------------------------------------------------
// Host orchestration.
// Input order = JAX pytree (sorted dict keys) flattening of setup_inputs():
// 0 x_a, 1 x_c, 2 edge_c, 3 edge_a, 4 ap_paper, 5 ap_author, 6 edge_lp,
// params: agg(W,b)=7,8; auth[0..2]{Wl,Wr,att,bias,bl,br}=9..26;
// cit[0..3]{...}=27..50; hi=51..54; if=55..58; lstm{Whh,Wih,bhh,bih}=59..62;
// num=63..66; post[0..4](W,b)=67..76; pre(W,b)=77,78; sjr=79..82.
// ---------------------------------------------------------------------------
extern "C" void kernel_launch(void* const* d_in, const int* in_sizes, int n_in,
                              void* d_out, int out_size, void* d_ws, size_t ws_size,
                              hipStream_t stream) {
  (void)in_sizes; (void)n_in; (void)out_size; (void)ws_size;
  auto F = [&](int i) { return (const float*)d_in[i]; };
  auto I = [&](int i) { return (const int*)d_in[i]; };

  const float* x_a = F(0);
  const float* x_c = F(1);
  const int* ec_src = I(2);           const int* ec_dst = I(2) + EC_;
  const int* ea_src = I(3);           const int* ea_dst = I(3) + EA_;
  const int* ap_paper = I(4);         const int* ap_author = I(5);
  const int* lp_src = I(6);           const int* lp_dst = I(6) + ELP_;
  float* out = (float*)d_out;

  // ---- workspace bump allocator ----
  char* cur = (char*)d_ws;
  auto alloc = [&](size_t bytes) -> char* {
    char* p = cur; cur += (bytes + 255) & ~(size_t)255; return p;
  };
  unsigned short* hbA   = (unsigned short*)alloc((size_t)NP_ * 128 * 2); // act bf16 (reused)
  unsigned short* hbB   = (unsigned short*)alloc((size_t)NA_ * 128 * 2); // act bf16 #2
  unsigned short* wb1   = (unsigned short*)alloc((size_t)128 * 512 * 2); // weight bf16
  unsigned short* wb2   = (unsigned short*)alloc((size_t)128 * 512 * 2);
  float* xl    = (float*)alloc((size_t)NP_ * 128 * 4);
  float* xr    = (float*)alloc((size_t)NP_ * 128 * 4);
  float* hp0   = (float*)alloc((size_t)NP_ * 128 * 4);
  float* hp1   = (float*)alloc((size_t)NP_ * 128 * 4);
  float* esc   = (float*)alloc((size_t)(EC_ + NP_) * 4);
  float* eex   = (float*)alloc((size_t)(EC_ + NP_) * 4);
  float* emax  = (float*)alloc((size_t)NP_ * 4);
  float* denom = (float*)alloc((size_t)NP_ * 4);
  float* collab = (float*)alloc((size_t)NA_ * 128 * 4);
  float* xcat   = (float*)alloc((size_t)NA_ * 147 * 4);
  unsigned short* xcat_b = (unsigned short*)alloc((size_t)NA_ * 160 * 2);
  float* aggout = (float*)alloc((size_t)NA_ * 147 * 4);
  float* conv[4];
  for (int i = 0; i < 4; ++i) conv[i] = (float*)alloc((size_t)NA_ * 128 * 4);
  float* xW   = (float*)alloc((size_t)NA_ * 512 * 4);
  float* ghb  = (float*)alloc((size_t)NA_ * 512 * 4);
  float* cbuf = (float*)alloc((size_t)NA_ * 128 * 4);
  float* hbuf = (float*)alloc((size_t)NA_ * 128 * 4);
  float* t1   = (float*)alloc((size_t)NA_ * 128 * 4);
  float* zhead[4];
  for (int i = 0; i < 4; ++i) zhead[i] = (float*)alloc((size_t)NA_ * 128 * 4);

  // ---- helpers ----
  auto fill = [&](float* p, float v, size_t n) {
    fill_f32_kernel<<<(unsigned)((n + 255) / 256), 256, 0, stream>>>(p, v, (long)n);
  };
  auto cvt = [&](const float* s, unsigned short* d, long rows, int K, int Kp) {
    long tot = rows * (long)Kp;
    cvt_bf16_kernel<<<(unsigned)((tot + 255) / 256), 256, 0, stream>>>(s, d, rows, K, Kp);
  };
  auto gemm = [&](const unsigned short* A, int Kp, const unsigned short* B, int ldb,
                  int K, const float* bias, float* C, int M, int N, int relu) {
    dim3 g((unsigned)((N + 127) / 128), (unsigned)((M + 127) / 128));
    wmma_gemm_kernel<<<g, 256, 0, stream>>>(A, Kp, B, ldb, K, bias, C, N, M, N, relu);
  };

  auto gat_layer = [&](const float* h_in, float* h_out, int Nn,
                       const int* srcp, const int* dstp, int E, int pbase) {
    const float* Wl = F(pbase + 0), *Wr = F(pbase + 1), *att = F(pbase + 2);
    const float* bias2 = F(pbase + 3), *bl = F(pbase + 4), *br = F(pbase + 5);
    cvt(h_in, hbA, Nn, 128, 128);
    cvt(Wl, wb1, 128, 128, 128);
    cvt(Wr, wb2, 128, 128, 128);
    gemm(hbA, 128, wb1, 128, 128, bl, xl, Nn, 128, 0);
    gemm(hbA, 128, wb2, 128, 128, br, xr, Nn, 128, 0);
    fill(emax, -1e30f, (size_t)Nn);
    fill(denom, 0.f, (size_t)Nn);
    fill(h_out, 0.f, (size_t)Nn * 128);
    int tot = E + Nn;
    gat_edge_score_kernel<<<(unsigned)((tot + 7) / 8), 256, 0, stream>>>(
        xl, xr, att, srcp, dstp, E, Nn, esc, emax);
    gat_edge_exp_kernel<<<(unsigned)((tot + 255) / 256), 256, 0, stream>>>(
        esc, emax, dstp, E, Nn, eex, denom);
    gat_scatter_kernel<<<(unsigned)((tot + 7) / 8), 256, 0, stream>>>(
        xl, eex, denom, srcp, dstp, E, Nn, h_out);
    bias_act_kernel<<<(unsigned)(((size_t)Nn * 128 + 255) / 256), 256, 0, stream>>>(
        h_out, bias2, (long)Nn, 128, 1);
  };

  // ---- citation GATv2 stack (4 layers) ----
  gat_layer(x_c, hp0, NP_, ec_src, ec_dst, EC_, 27 + 0 * 6);
  gat_layer(hp0, hp1, NP_, ec_src, ec_dst, EC_, 27 + 1 * 6);
  gat_layer(hp1, hp0, NP_, ec_src, ec_dst, EC_, 27 + 2 * 6);
  gat_layer(hp0, hp1, NP_, ec_src, ec_dst, EC_, 27 + 3 * 6);   // final paper emb

  // ---- per-author segment sum of paper embeddings ----
  fill(collab, 0.f, (size_t)NA_ * 128);
  segsum_kernel<<<(unsigned)((EAP_ + 7) / 8), 256, 0, stream>>>(
      hp1, ap_paper, ap_author, EAP_, collab);

  // ---- agg (147->147, relu) + pre (147->128) ----
  concat_kernel<<<(unsigned)(((size_t)NA_ * 147 + 255) / 256), 256, 0, stream>>>(
      x_a, collab, xcat, NA_);
  cvt(xcat, xcat_b, NA_, 147, 160);
  cvt(F(7), wb1, 147, 147, 147);
  gemm(xcat_b, 160, wb1, 147, 147, F(8), aggout, NA_, 147, 1);
  cvt(aggout, xcat_b, NA_, 147, 160);
  cvt(F(77), wb1, 147, 128, 128);
  gemm(xcat_b, 160, wb1, 128, 147, F(78), conv[0], NA_, 128, 0);

  // ---- author GATv2 stack (3 layers) ----
  gat_layer(conv[0], conv[1], NA_, ea_src, ea_dst, EA_, 9 + 0 * 6);
  gat_layer(conv[1], conv[2], NA_, ea_src, ea_dst, EA_, 9 + 1 * 6);
  gat_layer(conv[2], conv[3], NA_, ea_src, ea_dst, EA_, 9 + 2 * 6);

  // ---- LSTM over 4-step sequence of conv outputs ----
  const float* bhh = F(61), *bih = F(62);
  cvt(F(60), wb1, 128, 512, 512);   // Wih
  cvt(F(59), wb2, 128, 512, 512);   // Whh
  fill(hbuf, 0.f, (size_t)NA_ * 128);
  fill(cbuf, 0.f, (size_t)NA_ * 128);
  for (int t = 0; t < 4; ++t) {
    cvt(conv[t], hbA, NA_, 128, 128);
    gemm(hbA, 128, wb1, 512, 128, bih, xW, NA_, 512, 0);
    cvt(hbuf, hbB, NA_, 128, 128);
    gemm(hbB, 128, wb2, 512, 128, bhh, ghb, NA_, 512, 0);
    lstm_cell_kernel<<<(unsigned)(((size_t)NA_ * 128 + 255) / 256), 256, 0, stream>>>(
        xW, ghb, cbuf, hbuf, NA_);
  }

  // ---- four 2-layer heads: if(55), hi(51), sjr(79), num(63) ----
  cvt(hbuf, hbB, NA_, 128, 128);
  const int headBase[4] = {55, 51, 79, 63};    // zhead: 0=if, 1=hi, 2=sjr, 3=num
  for (int h = 0; h < 4; ++h) {
    int pb = headBase[h];
    cvt(F(pb + 0), wb1, 128, 128, 128);
    gemm(hbB, 128, wb1, 128, 128, F(pb + 1), t1, NA_, 128, 0);
    cvt(t1, hbA, NA_, 128, 128);
    cvt(F(pb + 2), wb2, 128, 128, 128);
    gemm(hbA, 128, wb2, 128, 128, F(pb + 3), zhead[h], NA_, 128, 0);
  }

  // ---- link prediction: embs = [had(x), had(sjr), had(hi), had(if), had(num)]
  //      with post[0..4] ----
  const float* zlist[5] = {hbuf, zhead[2], zhead[1], zhead[0], zhead[3]};
  for (int j = 0; j < 5; ++j) {
    link_head_kernel<<<(unsigned)((ELP_ + 7) / 8), 256, 0, stream>>>(
        zlist[j], lp_src, lp_dst, F(67 + 2 * j), F(68 + 2 * j),
        out + (size_t)j * ELP_, ELP_);
  }
}
